// LSTMDecoder_16372415332770
// MI455X (gfx1250) — matrix operand order repaired and checked by reading, not compiled, not run on hardware
//
#include <hip/hip_runtime.h>
#include <hip/hip_bf16.h>
#include <stdint.h>

// ---------------- problem constants (match reference) ----------------
#define BB   16
#define TT   64
#define DD   512
#define HH   512
#define NUP  256
#define NDN  256
#define VV   32000
#define OOVV 50
#define VO   (VV + OOVV)

// ---------------- CDNA5 WMMA types ----------------
typedef __attribute__((ext_vector_type(16))) __bf16          bf16x16;
typedef __attribute__((ext_vector_type(8)))  float           v8f;
typedef __attribute__((ext_vector_type(16))) unsigned short  u16x16;
typedef __attribute__((ext_vector_type(8)))  unsigned short  u16x8;

__device__ __forceinline__ unsigned short f2bf(float f) {
    // round-to-nearest-even fp32 -> bf16
    uint32_t x = __float_as_uint(f);
    uint32_t r = x + 0x7FFFu + ((x >> 16) & 1u);
    return (unsigned short)(r >> 16);
}

__device__ __forceinline__ float sigf(float x) { return 1.f / (1.f + expf(-x)); }

// ---- fragment loaders / compute group for the WMMA GEMM ----
__device__ __forceinline__ u16x16 load_a_frag(const unsigned short* apb, int k) {
    u16x8 lo = *(const u16x8*)(apb + k);
    u16x8 hi = *(const u16x8*)(apb + k + 8);
    return __builtin_shufflevector(lo, hi, 0,1,2,3,4,5,6,7,8,9,10,11,12,13,14,15);
}

__device__ __forceinline__ void load_b_frag4(u16x16 (&bu)[4],
                                             const unsigned short* const (&bp)[4], int k) {
#pragma unroll
    for (int j = 0; j < 4; ++j) bu[j] = *(const u16x16*)(bp[j] + k);
}

__device__ __forceinline__ void wmma_group4(v8f (&acc)[4], const u16x16& au,
                                            const u16x16 (&bu)[4]) {
    const bf16x16 a = __builtin_bit_cast(bf16x16, au);
#pragma unroll
    for (int j = 0; j < 4; ++j) {
        bf16x16 b = __builtin_bit_cast(bf16x16, bu[j]);
        acc[j] = __builtin_amdgcn_wmma_f32_16x16x32_bf16(false, a, false, b, (short)0,
                                                         acc[j], false, false);
    }
}

// ---------------- WMMA GEMM:  C[m,n] = sum_k A[m,k]*W[n,k] (+bias[n]) (+C if accum)
// A, W: bf16 row-major ([M,lda] / [N,ldb]). One wave owns a 1x4 strip of 16x16
// tiles. 2-stage software pipeline over K (requires K % 64 == 0 and K >= 64 --
// true for all call sites: K in {512,1536}). Each wmma group consumes fragments
// loaded a full stage earlier, so loads clause together and stay outstanding
// across the matrix ops.
__global__ void gemm_wmma_kernel(const unsigned short* __restrict__ Abf, int lda,
                                 const unsigned short* __restrict__ Bw, int ldb,
                                 const float* __restrict__ bias,
                                 float* __restrict__ C, int ldc,
                                 unsigned short* __restrict__ Cbf,
                                 int M, int N, int K, int accum) {
    const int lane   = threadIdx.x & 31;
    const int ntN    = N >> 4;
    const int groups = ntN >> 2;                       // 4 N-tiles per wave
    const int w      = blockIdx.x * (blockDim.x >> 5) + (threadIdx.x >> 5);
    if (w >= (M >> 4) * groups) return;                // wave-uniform: EXEC all-1 for WMMA
    const int tm  = w / groups;
    const int tn0 = (w % groups) << 2;

    const int row_a = (tm << 4) + (lane & 15);
    const int kbA   = (lane >> 4) << 4;                // A k order here (contiguous, same as B): hi lanes K+16
    const int kbB   = (lane >> 4) << 4;                // B k order here (contiguous, same as A): hi lanes K+16
    const int crow0 = (tm << 4) + ((lane >> 4) << 3);  // C layout: VGPR e <-> row crow0+e
    int coln[4];
#pragma unroll
    for (int j = 0; j < 4; ++j) coln[j] = ((tn0 + j) << 4) + (lane & 15);

    const unsigned short* apb = Abf + (size_t)row_a * lda + kbA;
    const unsigned short* bp[4];
#pragma unroll
    for (int j = 0; j < 4; ++j) bp[j] = Bw + (size_t)coln[j] * ldb + kbB;

    v8f acc[4];
    if (accum) {
#pragma unroll
        for (int j = 0; j < 4; ++j)
#pragma unroll
            for (int e = 0; e < 8; ++e) acc[j][e] = C[(size_t)(crow0 + e) * ldc + coln[j]];
    } else {
#pragma unroll
        for (int j = 0; j < 4; ++j)
#pragma unroll
            for (int e = 0; e < 8; ++e) acc[j][e] = 0.f;
    }

    // ---- 2-stage software pipeline over K (K % 64 == 0, K >= 64) ----
    u16x16 auA = load_a_frag(apb, 0);
    u16x16 buA[4]; load_b_frag4(buA, bp, 0);
    u16x16 auB = load_a_frag(apb, 32);
    u16x16 buB[4]; load_b_frag4(buB, bp, 32);

    for (int k = 64; k < K; k += 64) {
        wmma_group4(acc, auA, buA);
        auA = load_a_frag(apb, k);
        load_b_frag4(buA, bp, k);
        wmma_group4(acc, auB, buB);
        auB = load_a_frag(apb, k + 32);
        load_b_frag4(buB, bp, k + 32);
    }
    wmma_group4(acc, auA, buA);
    wmma_group4(acc, auB, buB);

#pragma unroll
    for (int j = 0; j < 4; ++j) {
        const float bb = bias ? bias[coln[j]] : 0.f;
#pragma unroll
        for (int e = 0; e < 8; ++e) {
            float v = acc[j][e] + bb;
            C[(size_t)(crow0 + e) * ldc + coln[j]] = v;
            if (Cbf) Cbf[(size_t)(crow0 + e) * ldc + coln[j]] = f2bf(v);
        }
    }
}

// ---------------- fp32 -> bf16 conversion ----------------
__global__ void cvt_bf16_kernel(const float* __restrict__ src,
                                unsigned short* __restrict__ dst, int n) {
    int i = blockIdx.x * blockDim.x + threadIdx.x;
    int stride = gridDim.x * blockDim.x;
    for (; i < n; i += stride) dst[i] = f2bf(src[i]);
}

// ---------------- initial state ----------------
__global__ void init_state_kernel(const float* __restrict__ h0, const float* __restrict__ c0,
                                  float* __restrict__ h, float* __restrict__ c,
                                  unsigned short* __restrict__ hbf) {
    int i = blockIdx.x * blockDim.x + threadIdx.x;
    if (i < BB * HH) { float v = h0[i]; h[i] = v; hbf[i] = f2bf(v); c[i] = c0[i]; }
}

// ---------------- Bahdanau attend: scores + softmax + context ----------------
// block = one batch row, 256 threads (8 wave32). N = 256 nodes, H = 512.
__global__ void attend_kernel(const float* __restrict__ q,    // [B,H]
                              const float* __restrict__ kp,   // [B,N,H]
                              const float* __restrict__ keys, // [B,N,H]
                              const float* __restrict__ mask, // [B,N]
                              const float* __restrict__ vvec, // [H]
                              const float* __restrict__ bv,   // [1]
                              float* __restrict__ ctx,        // [B,H] or null
                              float* __restrict__ aout) {     // [B,N] or null
    const int b = blockIdx.x, tid = threadIdx.x;
    const int lane = tid & 31, warp = tid >> 5;
    __shared__ float s_s[NDN];
    __shared__ float red[256];
    const float* qb = q + (size_t)b * HH;

    for (int n = warp; n < NDN; n += 8) {
        const float* kpn = kp + ((size_t)b * NDN + n) * HH;
        float part = 0.f;
        for (int j = lane; j < HH; j += 32) part += vvec[j] * tanhf(qb[j] + kpn[j]);
#pragma unroll
        for (int off = 16; off; off >>= 1) part += __shfl_xor(part, off);
        if (lane == 0) {
            float s = part + bv[0];
            s_s[n] = (mask[(size_t)b * NDN + n] > 0.f) ? s : -1e9f;
        }
    }
    __syncthreads();

    float sv = s_s[tid];
    red[tid] = sv; __syncthreads();
#pragma unroll
    for (int off = 128; off; off >>= 1) { if (tid < off) red[tid] = fmaxf(red[tid], red[tid + off]); __syncthreads(); }
    float m = red[0]; __syncthreads();
    float e = expf(sv - m);
    red[tid] = e; __syncthreads();
#pragma unroll
    for (int off = 128; off; off >>= 1) { if (tid < off) red[tid] += red[tid + off]; __syncthreads(); }
    float a = e / red[0];
    __syncthreads();
    s_s[tid] = a;
    __syncthreads();

    if (ctx) {
        float a0 = 0.f, a1 = 0.f;
        for (int n = 0; n < NDN; ++n) {
            float an = s_s[n];
            const float* kn = keys + ((size_t)b * NDN + n) * HH;
            a0 += an * kn[tid];
            a1 += an * kn[tid + 256];
        }
        ctx[(size_t)b * HH + tid]       = a0;
        ctx[(size_t)b * HH + tid + 256] = a1;
    }
    if (aout) aout[(size_t)b * NDN + tid] = a;
}

// ---------------- switch probability p = sigmoid(h . Wsw + bsw) ----------------
__global__ void switch_kernel(const float* __restrict__ h, const float* __restrict__ Wsw,
                              const float* __restrict__ bsw, float* __restrict__ pgen) {
    const int b = blockIdx.x, lane = threadIdx.x;
    float part = 0.f;
    for (int j = lane; j < HH; j += 32) part += Wsw[j] * h[(size_t)b * HH + j];
#pragma unroll
    for (int off = 16; off; off >>= 1) part += __shfl_xor(part, off);
    if (lane == 0) pgen[b] = sigf(part + bsw[0]);
}

// ---------------- build concat [q_t, up_ctx, dn_ctx] as bf16 ----------------
__global__ void build_xcat_kernel(const float* __restrict__ questions, int t,
                                  const float* __restrict__ ctx_up,
                                  const float* __restrict__ ctx_dn,
                                  unsigned short* __restrict__ xcat) {
    int i = blockIdx.x * blockDim.x + threadIdx.x;
    const int W = DD + 2 * HH;                  // 1536
    if (i >= BB * W) return;
    int b = i / W, j = i % W;
    float v;
    if (j < DD)            v = questions[((size_t)b * TT + t) * DD + j];
    else if (j < DD + HH)  v = ctx_up[(size_t)b * HH + (j - DD)];
    else                   v = ctx_dn[(size_t)b * HH + (j - DD - HH)];
    xcat[i] = f2bf(v);
}

// ---------------- LSTM pointwise ----------------
__global__ void lstm_pointwise_kernel(const float* __restrict__ gates,
                                      const float* __restrict__ c_in,
                                      float* __restrict__ h_out, float* __restrict__ c_out,
                                      unsigned short* __restrict__ hbf_out) {
    int i = blockIdx.x * blockDim.x + threadIdx.x;
    if (i >= BB * HH) return;
    int b = i / HH, j = i % HH;
    const float* g = gates + (size_t)b * 4 * HH;
    float ig = sigf(g[j]);
    float fg = sigf(g[HH + j]);
    float gg = tanhf(g[2 * HH + j]);
    float og = sigf(g[3 * HH + j]);
    float c2 = fg * c_in[i] + ig * gg;
    float h2 = og * tanhf(c2);
    c_out[i] = c2;
    h_out[i] = h2;
    hbf_out[i] = f2bf(h2);
}

// ---------------- vocab softmax, scaled by p, written straight to out ----------------
__global__ void softmax_fill_kernel(const float* __restrict__ logits,
                                    const float* __restrict__ pgen,
                                    float* __restrict__ out, int t) {
    const int b = blockIdx.x, tid = threadIdx.x;
    __shared__ float red[256];
    const float* lg = logits + (size_t)b * VV;
    float m = -1e30f;
    for (int v = tid; v < VV; v += 256) m = fmaxf(m, lg[v]);
    red[tid] = m; __syncthreads();
#pragma unroll
    for (int off = 128; off; off >>= 1) { if (tid < off) red[tid] = fmaxf(red[tid], red[tid + off]); __syncthreads(); }
    m = red[0]; __syncthreads();
    float s = 0.f;
    for (int v = tid; v < VV; v += 256) s += expf(lg[v] - m);
    red[tid] = s; __syncthreads();
#pragma unroll
    for (int off = 128; off; off >>= 1) { if (tid < off) red[tid] += red[tid + off]; __syncthreads(); }
    float inv = pgen[b] / red[0];
    float* orow = out + ((size_t)b * TT + t) * (size_t)VO;
    for (int v = tid; v < VV; v += 256) orow[v] = expf(lg[v] - m) * inv;
    for (int v = VV + tid; v < VO; v += 256) orow[v] = 0.f;
}

// ---------------- copy-distribution scatter add ----------------
__global__ void scatter_copy_kernel(const float* __restrict__ ca,
                                    const float* __restrict__ pgen,
                                    const int* __restrict__ map,
                                    float* __restrict__ out, int t) {
    const int b = blockIdx.x, n = threadIdx.x;
    float w = (1.f - pgen[b]) * ca[(size_t)b * NDN + n];
    int idx = map[b * NDN + n];
    atomicAdd(out + ((size_t)b * TT + t) * (size_t)VO + idx, w);
}

// ---------------- final h/c copy ----------------
__global__ void copy_final_kernel(const float* __restrict__ h, const float* __restrict__ c,
                                  float* __restrict__ out) {
    int i = blockIdx.x * blockDim.x + threadIdx.x;
    if (i < BB * HH) {
        out[(size_t)BB * TT * VO + i] = h[i];
        out[(size_t)BB * TT * VO + BB * HH + i] = c[i];
    }
}

// =====================================================================
extern "C" void kernel_launch(void* const* d_in, const int* in_sizes, int n_in,
                              void* d_out, int out_size, void* d_ws, size_t ws_size,
                              hipStream_t stream) {
    (void)in_sizes; (void)n_in; (void)out_size; (void)ws_size;

    const float* questions  = (const float*)d_in[0];
    const float* h0         = (const float*)d_in[1];
    const float* c0         = (const float*)d_in[2];
    const float* up_nodes   = (const float*)d_in[3];
    const float* down_nodes = (const float*)d_in[4];
    const float* up_mask    = (const float*)d_in[5];
    const float* down_mask  = (const float*)d_in[6];
    const float* copy_mask  = (const float*)d_in[7];
    const int*   src2trg    = (const int*)d_in[8];
    const float* Wq_up = (const float*)d_in[9],  *bq_up = (const float*)d_in[10];
    const float* Wk_up = (const float*)d_in[11], *bk_up = (const float*)d_in[12];
    const float* v_up  = (const float*)d_in[13], *bv_up = (const float*)d_in[14];
    const float* Wq_dn = (const float*)d_in[15], *bq_dn = (const float*)d_in[16];
    const float* Wk_dn = (const float*)d_in[17], *bk_dn = (const float*)d_in[18];
    const float* v_dn  = (const float*)d_in[19], *bv_dn = (const float*)d_in[20];
    const float* Wq_cp = (const float*)d_in[21], *bq_cp = (const float*)d_in[22];
    const float* Wk_cp = (const float*)d_in[23], *bk_cp = (const float*)d_in[24];
    const float* v_cp  = (const float*)d_in[25], *bv_cp = (const float*)d_in[26];
    const float* Wih  = (const float*)d_in[27], *Whh  = (const float*)d_in[28];
    const float* bih  = (const float*)d_in[29], *bhh  = (const float*)d_in[30];
    const float* Wprj = (const float*)d_in[31], *bprj = (const float*)d_in[32];
    const float* Wout = (const float*)d_in[33], *bout = (const float*)d_in[34];
    const float* Wsw  = (const float*)d_in[35], *bsw  = (const float*)d_in[36];

    float* out = (float*)d_out;

    // ---- workspace carve (~78 MB) ----
    char* wp = (char*)d_ws;
    auto carve = [&](size_t bytes) -> char* {
        char* r = wp;
        wp += (bytes + 255) & ~(size_t)255;
        return r;
    };
    unsigned short* Wout_bf = (unsigned short*)carve(2ull * VV * HH);
    unsigned short* Wih_bf  = (unsigned short*)carve(2ull * 4 * HH * HH);
    unsigned short* Whh_bf  = (unsigned short*)carve(2ull * 4 * HH * HH);
    unsigned short* Wprj_bf = (unsigned short*)carve(2ull * HH * (DD + 2 * HH));
    unsigned short* Wq_bf[3], *Wk_bf[3];
    for (int i = 0; i < 3; ++i) Wq_bf[i] = (unsigned short*)carve(2ull * HH * HH);
    for (int i = 0; i < 3; ++i) Wk_bf[i] = (unsigned short*)carve(2ull * HH * HH);
    unsigned short* upn_bf = (unsigned short*)carve(2ull * BB * NUP * HH);
    unsigned short* dnn_bf = (unsigned short*)carve(2ull * BB * NDN * HH);
    float* up_kp = (float*)carve(4ull * BB * NUP * HH);
    float* dn_kp = (float*)carve(4ull * BB * NDN * HH);
    float* cp_kp = (float*)carve(4ull * BB * NDN * HH);
    float* hbuf[2], *cbuf[2];
    unsigned short* hbf[2];
    for (int i = 0; i < 2; ++i) hbuf[i] = (float*)carve(4ull * BB * HH);
    for (int i = 0; i < 2; ++i) cbuf[i] = (float*)carve(4ull * BB * HH);
    for (int i = 0; i < 2; ++i) hbf[i]  = (unsigned short*)carve(2ull * BB * HH);
    float* q_up   = (float*)carve(4ull * BB * HH);
    float* q_dn   = (float*)carve(4ull * BB * HH);
    float* q_cp   = (float*)carve(4ull * BB * HH);
    float* ctx_up = (float*)carve(4ull * BB * HH);
    float* ctx_dn = (float*)carve(4ull * BB * HH);
    float* ca     = (float*)carve(4ull * BB * NDN);
    float* pgen   = (float*)carve(4ull * BB);
    unsigned short* xcat_bf = (unsigned short*)carve(2ull * BB * (DD + 2 * HH));
    float* xp     = (float*)carve(4ull * BB * HH);
    unsigned short* xp_bf = (unsigned short*)carve(2ull * BB * HH);
    float* gates  = (float*)carve(4ull * BB * 4 * HH);
    float* logits = (float*)carve(4ull * BB * VV);

    auto cvt = [&](const float* s, unsigned short* d, int n) {
        int blocks = (n + 255) / 256;
        if (blocks > 4096) blocks = 4096;
        cvt_bf16_kernel<<<blocks, 256, 0, stream>>>(s, d, n);
    };
    // C[m,n] = sum_k A[m,k]*W[n,k] (+bias) (+C if accum); each wave owns 4 N-tiles
    auto gemm = [&](const unsigned short* Abf, int lda,
                    const unsigned short* Bw, int ldb, const float* bias,
                    float* C, int ldc, unsigned short* Cbf,
                    int M, int N, int K, int accum) {
        int tiles = (M / 16) * (N / 64);        // wave strips of 4 N-tiles
        int blocks = (tiles + 3) / 4;           // 4 wave32 per block
        gemm_wmma_kernel<<<blocks, 128, 0, stream>>>(Abf, lda, Bw, ldb, bias,
                                                     C, ldc, Cbf, M, N, K, accum);
    };

    // ---- one-time: weights + attention keys -> bf16 ----
    cvt(Wout, Wout_bf, VV * HH);
    cvt(Wih,  Wih_bf,  4 * HH * HH);
    cvt(Whh,  Whh_bf,  4 * HH * HH);
    cvt(Wprj, Wprj_bf, HH * (DD + 2 * HH));
    cvt(Wq_up, Wq_bf[0], HH * HH); cvt(Wq_dn, Wq_bf[1], HH * HH); cvt(Wq_cp, Wq_bf[2], HH * HH);
    cvt(Wk_up, Wk_bf[0], HH * HH); cvt(Wk_dn, Wk_bf[1], HH * HH); cvt(Wk_cp, Wk_bf[2], HH * HH);
    cvt(up_nodes,   upn_bf, BB * NUP * HH);
    cvt(down_nodes, dnn_bf, BB * NDN * HH);

    // ---- one-time: hoisted key projections kp = nodes @ Wk.T + bk ----
    gemm(upn_bf, HH, Wk_bf[0], HH, bk_up, up_kp, HH, nullptr, BB * NUP, HH, HH, 0);
    gemm(dnn_bf, HH, Wk_bf[1], HH, bk_dn, dn_kp, HH, nullptr, BB * NDN, HH, HH, 0);
    gemm(dnn_bf, HH, Wk_bf[2], HH, bk_cp, cp_kp, HH, nullptr, BB * NDN, HH, HH, 0);

    // ---- initial state ----
    init_state_kernel<<<(BB * HH + 255) / 256, 256, 0, stream>>>(h0, c0, hbuf[0], cbuf[0], hbf[0]);

    // ---- sequential decode ----
    for (int t = 0; t < TT; ++t) {
        const int in = t & 1, on = in ^ 1;
        const float* h_in = hbuf[in];
        const unsigned short* hbf_in = hbf[in];

        // q projections from PRE-step h
        gemm(hbf_in, HH, Wq_bf[0], HH, bq_up, q_up, HH, nullptr, BB, HH, HH, 0);
        gemm(hbf_in, HH, Wq_bf[1], HH, bq_dn, q_dn, HH, nullptr, BB, HH, HH, 0);
        gemm(hbf_in, HH, Wq_bf[2], HH, bq_cp, q_cp, HH, nullptr, BB, HH, HH, 0);

        attend_kernel<<<BB, 256, 0, stream>>>(q_up, up_kp, up_nodes, up_mask, v_up, bv_up,
                                              ctx_up, nullptr);
        attend_kernel<<<BB, 256, 0, stream>>>(q_dn, dn_kp, down_nodes, down_mask, v_dn, bv_dn,
                                              ctx_dn, nullptr);
        attend_kernel<<<BB, 256, 0, stream>>>(q_cp, cp_kp, down_nodes, copy_mask, v_cp, bv_cp,
                                              nullptr, ca);

        switch_kernel<<<BB, 32, 0, stream>>>(h_in, Wsw, bsw, pgen);

        build_xcat_kernel<<<(BB * (DD + 2 * HH) + 255) / 256, 256, 0, stream>>>(
            questions, t, ctx_up, ctx_dn, xcat_bf);

        // x = concat @ Wprj.T + bprj
        gemm(xcat_bf, DD + 2 * HH, Wprj_bf, DD + 2 * HH, bprj,
             xp, HH, xp_bf, BB, HH, DD + 2 * HH, 0);

        // gates = x @ Wih.T + bih + h @ Whh.T + bhh
        gemm(xp_bf,  HH, Wih_bf, HH, bih, gates, 4 * HH, nullptr, BB, 4 * HH, HH, 0);
        gemm(hbf_in, HH, Whh_bf, HH, bhh, gates, 4 * HH, nullptr, BB, 4 * HH, HH, 1);

        lstm_pointwise_kernel<<<(BB * HH + 255) / 256, 256, 0, stream>>>(
            gates, cbuf[in], hbuf[on], cbuf[on], hbf[on]);

        // logits = h2 @ Wout.T + bout   (16 x 32000 x 512, dominant WMMA GEMM)
        gemm(hbf[on], HH, Wout_bf, HH, bout, logits, VV, nullptr, BB, VV, HH, 0);

        softmax_fill_kernel<<<BB, 256, 0, stream>>>(logits, pgen, out, t);
        scatter_copy_kernel<<<BB, 256, 0, stream>>>(ca, pgen, src2trg, out, t);
    }

    // after t = 63 (odd), final state sits in buffer 0
    copy_final_kernel<<<(BB * HH + 255) / 256, 256, 0, stream>>>(hbuf[0], cbuf[0], out);
}
